// BNet_60601988547111
// MI455X (gfx1250) — matrix operand order repaired and checked by reading, not compiled, not run on hardware
//
#include <hip/hip_runtime.h>
#include <hip/hip_bf16.h>

typedef __attribute__((ext_vector_type(16))) _Float16 v16h;
typedef __attribute__((ext_vector_type(8)))  float    v8f;

constexpr int   NN   = 200000;   // nodes
constexpr int   NHE  = 100000;   // hyperedges
constexpr long  NNZp = 1000000;  // pins
constexpr int   CC   = 64;       // hidden channels

__device__ __forceinline__ float lrelu(float v) { return v >= 0.0f ? v : 0.1f * v; }

// order-preserving float <-> uint encoding (for atomic max/min on floats)
__device__ __forceinline__ unsigned encf(float f) {
    unsigned u = __float_as_uint(f);
    return (u & 0x80000000u) ? ~u : (u | 0x80000000u);
}
__device__ __forceinline__ float decf(unsigned u) {
    return __uint_as_float((u & 0x80000000u) ? (u & 0x7FFFFFFFu) : ~u);
}

// char-aliased half store (safe in-place f32 -> f16 repack)
__device__ __forceinline__ void store_half(void* base, long hidx, float v) {
    _Float16 h = (_Float16)v;
    __builtin_memcpy((char*)base + hidx * 2, &h, 2);
}

// ---------------- init fills ----------------
__global__ void fill_f32(float* p, long n, float v) {
    long i = (long)blockIdx.x * blockDim.x + threadIdx.x;
    long s = (long)gridDim.x * blockDim.x;
    for (; i < n; i += s) p[i] = v;
}
__global__ void fill_u32(unsigned* p, long n, unsigned v) {
    long i = (long)blockIdx.x * blockDim.x + threadIdx.x;
    long s = (long)gridDim.x * blockDim.x;
    for (; i < n; i += s) p[i] = v;
}

__global__ void scatter_ismacro(const int* __restrict__ midx, float* __restrict__ ism, int nm) {
    int i = blockIdx.x * blockDim.x + threadIdx.x;
    if (i < nm) ism[midx[i]] = 1.0f;
}

// WT[n*K + k] = (f16) W[k*N + n]   (transposed, fragment-friendly: contiguous in k)
__global__ void transpose_w_f16(const float* __restrict__ W, _Float16* __restrict__ WT,
                                int K, int N) {
    int i = blockIdx.x * blockDim.x + threadIdx.x;
    if (i >= K * N) return;
    int n = i / K, k = i - n * K;
    WT[i] = (_Float16)W[k * N + n];
}

// ---------------- xl = [x|fake_pos|ismacro] @ W1 + b1  (WMMA, K=32) ----------------
__global__ void xl_gemm(const float* __restrict__ x, const float* __restrict__ fpos,
                        const float* __restrict__ ism, const _Float16* __restrict__ W1T,
                        const float* __restrict__ b1, float* __restrict__ xl, int ntiles) {
    int lane = threadIdx.x & 31;
    int wv   = threadIdx.x >> 5;
    int tile = blockIdx.x * (blockDim.x >> 5) + wv;
    if (tile >= ntiles) return;
    int r   = lane & 15;
    int hi  = lane >> 4;              // 0 / 1
    long row = (long)tile * 16 + r;

    // A fragment: 16x32 f16 (row M = lane&15; K pairs per ISA layout)
    v16h a;
#pragma unroll
    for (int g = 0; g < 8; ++g) {
        int k0 = ((g < 4) ? (2 * g) : (16 + 2 * (g - 4))) + hi * 8;
#pragma unroll
        for (int h2 = 0; h2 < 2; ++h2) {
            int k = k0 + h2;
            float v;
            if (k < 29)      v = x[row * 29 + k];
            else if (k < 31) v = fpos[row * 2 + (k - 29)];
            else             v = ism[row];
            a[2 * g + h2] = (_Float16)v;
        }
    }

    int col0 = lane & 15;
#pragma unroll
    for (int t = 0; t < 4; ++t) {
        int c = t * 16 + col0;
        // B fragment: lane needs W1[k][c] for k = hi*8 .. hi*8+7 and 16+hi*8 .. 16+hi*8+7 (table order)
        // -> two 16-byte runs of W1T row c
        v16h b;
        __builtin_memcpy(&b, &W1T[c * 32 + hi * 8], 16);
        __builtin_memcpy((char*)&b + 16, &W1T[c * 32 + 16 + hi * 8], 16);
        float bv = b1[c];
        v8f acc;
#pragma unroll
        for (int j = 0; j < 8; ++j) acc[j] = bv;
        acc = __builtin_amdgcn_wmma_f32_16x16x32_f16(false, a, false, b, (short)0, acc, false, false);
#pragma unroll
        for (int j = 0; j < 8; ++j) {
            int m = j + hi * 8;
            xl[((long)tile * 16 + m) * CC + c] = acc[j];
        }
    }
}

// ---------------- pin -> hyperedge scatter: efeat += xl[node] + pin@Wpin ----------------
__global__ void pin_scatter_he(const int* __restrict__ ni, const int* __restrict__ ei,
                               const float* __restrict__ pinf, const float* __restrict__ Wpin,
                               const float* __restrict__ xl, float* __restrict__ efeat,
                               float* __restrict__ dege) {
    long tid = (long)blockIdx.x * blockDim.x + threadIdx.x;
    if (tid >= NNZp * 16) return;
    int p = (int)(tid >> 4);
    int q = (int)(tid & 15);
    int n = ni[p], e = ei[p];
    float4 pf = ((const float4*)pinf)[p];
    int c0 = q * 4;
#pragma unroll
    for (int j = 0; j < 4; ++j) {
        int c = c0 + j;
        float pm = pf.x * Wpin[0 * CC + c] + pf.y * Wpin[1 * CC + c]
                 + pf.z * Wpin[2 * CC + c] + pf.w * Wpin[3 * CC + c];
        atomicAdd(&efeat[(long)e * CC + c], xl[(long)n * CC + c] + pm);
    }
    if (q == 0) atomicAdd(&dege[e], 1.0f);
}

// ---------------- e_feat /= max(deg_e, 1) ----------------
__global__ void he_norm(float* __restrict__ efeat, const float* __restrict__ dege) {
    long i = (long)blockIdx.x * blockDim.x + threadIdx.x;
    long s = (long)gridDim.x * blockDim.x;
    for (; i < (long)NHE * CC; i += s) {
        int e = (int)(i >> 6);
        efeat[i] /= fmaxf(dege[e], 1.0f);
    }
}

// ---------------- per-row dot with half of att ----------------
__global__ void dot_att(const float* __restrict__ feat, const float* __restrict__ att,
                        int attoff, float* __restrict__ out, int n) {
    int i = blockIdx.x * blockDim.x + threadIdx.x;
    if (i >= n) return;
    const float4* f4 = (const float4*)(feat + (long)i * CC);
    const float4* a4 = (const float4*)(att + attoff);
    float s = 0.0f;
#pragma unroll
    for (int q = 0; q < 16; ++q) {
        float4 f = f4[q], a = a4[q];
        s += f.x * a.x + f.y * a.y + f.z * a.z + f.w * a.w;
    }
    out[i] = s;
}

// ---------------- a_raw = lrelu(natt[node]+heatt[he]); segment max per node ----------------
__global__ void araw_max(const int* __restrict__ ni, const int* __restrict__ ei,
                         const float* __restrict__ natt, const float* __restrict__ heatt,
                         float* __restrict__ araw, unsigned* __restrict__ menc) {
    long p = (long)blockIdx.x * blockDim.x + threadIdx.x;
    if (p >= NNZp) return;
    float a = lrelu(natt[ni[p]] + heatt[ei[p]]);
    araw[p] = a;
    atomicMax(&menc[ni[p]], encf(a));
}

__global__ void decode_m(unsigned* __restrict__ menc, int n) {
    int i = blockIdx.x * blockDim.x + threadIdx.x;
    if (i >= n) return;
    float f = decf(menc[i]);
    if (!(f >= -3.0e38f && f <= 3.0e38f)) f = 0.0f;   // !isfinite -> 0
    menc[i] = __float_as_uint(f);                     // buffer now holds plain float m
}

// ---------------- ex = exp(a_raw - m); den += ex; deg_n += 1 ----------------
__global__ void exp_den(const int* __restrict__ ni, unsigned* __restrict__ mbuf,
                        float* __restrict__ araw, float* __restrict__ den,
                        float* __restrict__ degn) {
    long p = (long)blockIdx.x * blockDim.x + threadIdx.x;
    if (p >= NNZp) return;
    int n = ni[p];
    float m = __uint_as_float(mbuf[n]);
    float e = __expf(araw[p] - m);
    araw[p] = e;                        // in-place: araw buffer now holds ex
    atomicAdd(&den[n], e);
    atomicAdd(&degn[n], 1.0f);
}

// ---------------- msg = e_g * alpha; scatter sum/sumsq/max/min per node ----------------
__global__ void pna_scatter(const int* __restrict__ ni, const int* __restrict__ ei,
                            const float* __restrict__ exbuf, const float* __restrict__ den,
                            const float* __restrict__ efeat,
                            float* __restrict__ sumb, float* __restrict__ sqb,
                            unsigned* __restrict__ mxe, unsigned* __restrict__ mne) {
    long tid = (long)blockIdx.x * blockDim.x + threadIdx.x;
    if (tid >= NNZp * 16) return;
    int p = (int)(tid >> 4);
    int q = (int)(tid & 15);
    int n = ni[p], e = ei[p];
    float alpha = exbuf[p] / (den[n] + 1e-16f);
    float4 eg = ((const float4*)efeat)[(long)e * 16 + q];
    float ms[4] = { eg.x * alpha, eg.y * alpha, eg.z * alpha, eg.w * alpha };
#pragma unroll
    for (int j = 0; j < 4; ++j) {
        long i = (long)n * CC + q * 4 + j;
        atomicAdd(&sumb[i], ms[j]);
        atomicAdd(&sqb[i], ms[j] * ms[j]);
        atomicMax(&mxe[i], encf(ms[j]));
        atomicMin(&mne[i], encf(ms[j]));
    }
}

// ---------------- per-node stats -> f16 PNA features, repacked in-place ----------------
// sumb row (64 f32) -> halves [mean(0..63) | max(0..63)]  (exactly 128 f16)
// sqb  row (64 f32) -> halves [min (0..63) | std(0..63)]
__global__ void pna_stats(float* __restrict__ sumb, float* __restrict__ sqb,
                          const unsigned* __restrict__ mxe, const unsigned* __restrict__ mne,
                          const float* __restrict__ degn) {
    int n = blockIdx.x * blockDim.x + threadIdx.x;
    if (n >= NN) return;
    float deg = degn[n];
    float dc  = fmaxf(deg, 1.0f);
    bool  has = deg > 0.0f;
    float mxbuf[64], stbuf[64];
    for (int c = 0; c < 64; ++c) {
        long i = (long)n * CC + c;
        float s  = sumb[i];
        float ss = sqb[i];
        float mean = s / dc;
        float mx = has ? decf(mxe[i]) : 0.0f;
        float mn = has ? decf(mne[i]) : 0.0f;
        float var = fmaxf(ss / dc - mean * mean, 0.0f);
        float sd  = sqrtf(var + 1e-12f);
        // writes target f32 slot c/2 of the same row: slot already consumed.
        store_half(sumb, (long)n * 128 + c, mean);
        store_half(sqb,  (long)n * 128 + c, mn);
        mxbuf[c] = mx;
        stbuf[c] = sd;
    }
    for (int c = 0; c < 64; ++c) {
        store_half(sumb, (long)n * 128 + 64 + c, mxbuf[c]);
        store_half(sqb,  (long)n * 128 + 64 + c, stbuf[c]);
    }
}

// ---------------- hx = lrelu(pna(256) @ Wpost + bpost)  (WMMA, K=256 = 8x32) ----------------
// Wpost^T (f16, 32KB) staged once per block into LDS via CDNA5 async copy.
__global__ void hx_gemm(const void* __restrict__ pnaA, const void* __restrict__ pnaB,
                        const _Float16* __restrict__ WT, const float* __restrict__ bpost,
                        float* __restrict__ hx, int ntiles) {
    __shared__ _Float16 smWT[64 * 256];   // 32 KB, WT[c][k]

    // async stage: 256 threads x 8 iters x 16B = 32KB (ASYNCcnt tracked)
    {
        unsigned lds0   = (unsigned)(unsigned long long)(&smWT[0]) + threadIdx.x * 16u;
        const char* g0  = (const char*)WT + threadIdx.x * 16;
#pragma unroll
        for (int it = 0; it < 8; ++it) {
            unsigned   l = lds0 + it * 4096u;
            const char* g = g0 + it * 4096;
            asm volatile("global_load_async_to_lds_b128 %0, %1, off"
                         :: "v"(l), "v"(g) : "memory");
        }
        asm volatile("s_wait_asynccnt 0x0" ::: "memory");
    }
    __syncthreads();

    int lane = threadIdx.x & 31;
    int wv   = threadIdx.x >> 5;
    int tile = blockIdx.x * (blockDim.x >> 5) + wv;
    if (tile >= ntiles) return;
    int r    = lane & 15;
    int hi   = lane >> 4;
    long row = (long)tile * 16 + r;
    int col0 = lane & 15;

    v8f acc[4];
#pragma unroll
    for (int t = 0; t < 4; ++t) {
        float bv = bpost[t * 16 + col0];
#pragma unroll
        for (int j = 0; j < 8; ++j) acc[t][j] = bv;
    }

    const _Float16* pa = (const _Float16*)pnaA;
    const _Float16* pb = (const _Float16*)pnaB;

#pragma unroll
    for (int kk = 0; kk < 8; ++kk) {
        // A fragment: two contiguous 8-half (16B) runs of this row's PNA features
        const _Float16* src = (kk < 4) ? (pa + row * 128 + kk * 32)
                                       : (pb + row * 128 + (kk - 4) * 32);
        v16h a;
        __builtin_memcpy(&a, src + hi * 8, 16);
        __builtin_memcpy((char*)&a + 16, src + 16 + hi * 8, 16);
#pragma unroll
        for (int t = 0; t < 4; ++t) {
            int c = t * 16 + col0;
            // B fragment: two 16B runs of WT row c in LDS (k = hi*8.., 16+hi*8.., matching A)
            v16h b;
            __builtin_memcpy(&b, &smWT[c * 256 + kk * 32 + hi * 8], 16);
            __builtin_memcpy((char*)&b + 16, &smWT[c * 256 + kk * 32 + 16 + hi * 8], 16);
            acc[t] = __builtin_amdgcn_wmma_f32_16x16x32_f16(false, a, false, b, (short)0,
                                                            acc[t], false, false);
        }
    }
#pragma unroll
    for (int t = 0; t < 4; ++t)
#pragma unroll
        for (int j = 0; j < 8; ++j) {
            int m = j + hi * 8;
            hx[((long)tile * 16 + m) * CC + t * 16 + col0] = lrelu(acc[t][j]);
        }
}

// ---------------- pooling ----------------
__global__ void pool_all(const float* __restrict__ hx, float* __restrict__ pooled) {
    __shared__ float sm[256];
    int t = threadIdx.x;
    int c = t & 63, g = t >> 6;
    float local = 0.0f;
    for (long n = (long)blockIdx.x * 4 + g; n < NN; n += (long)gridDim.x * 4)
        local += hx[n * CC + c];
    sm[t] = local;
    __syncthreads();
    if (g == 0) atomicAdd(&pooled[64 + c], sm[c] + sm[64 + c] + sm[128 + c] + sm[192 + c]);
}

__global__ void pool_macro(const float* __restrict__ hx, const int* __restrict__ midx,
                           float* __restrict__ pooled) {
    __shared__ float sm[256];
    int t = threadIdx.x;
    int c = t & 63, g = t >> 6;
    float local = 0.0f;
    for (int i = g; i < 512; i += 4) local += hx[(long)midx[i] * CC + c];
    sm[t] = local;
    __syncthreads();
    if (g == 0) pooled[c] = sm[c] + sm[64 + c] + sm[128 + c] + sm[192 + c];
}

// ---------------- final MLP head (single block) ----------------
__global__ void mlp_head(const float* __restrict__ pooled,
                         const float* __restrict__ Wm1, const float* __restrict__ bm1,
                         const float* __restrict__ Wm2, const float* __restrict__ bm2,
                         const float* __restrict__ Wm3, const float* __restrict__ bm3,
                         float* __restrict__ out) {
    __shared__ float p[128];
    __shared__ float z1[64];
    __shared__ float z2[32];
    int t = threadIdx.x;
    if (t < 128) p[t] = pooled[t] * (t < 64 ? (1.0f / 512.0f) : (1.0f / 200000.0f));
    __syncthreads();
    if (t < 64) {
        float s = bm1[t];
        for (int k = 0; k < 128; ++k) s += p[k] * Wm1[k * 64 + t];
        z1[t] = lrelu(s);
    }
    __syncthreads();
    if (t < 32) {
        float s = bm2[t];
        for (int k = 0; k < 64; ++k) s += z1[k] * Wm2[k * 32 + t];
        z2[t] = lrelu(s);
    }
    __syncthreads();
    if (t == 0) {
        float s = bm3[0];
        for (int k = 0; k < 32; ++k) s += z2[k] * Wm3[k];
        out[0] = s;
    }
}

static inline int gridFor(long n, int blk) {
    long g = (n + blk - 1) / blk;
    return (int)(g > 8192 ? 8192 : g);
}

extern "C" void kernel_launch(void* const* d_in, const int* in_sizes, int n_in,
                              void* d_out, int out_size, void* d_ws, size_t ws_size,
                              hipStream_t stream) {
    const float* x     = (const float*)d_in[0];
    const float* fpos  = (const float*)d_in[1];
    const int*   eidx  = (const int*)d_in[2];
    const int*   ni    = eidx;              // edge_index[0]
    const int*   ei    = eidx + NNZp;       // edge_index[1]
    const float* pinf  = (const float*)d_in[3];
    const int*   midx  = (const int*)d_in[4];
    const float* W1    = (const float*)d_in[5];
    const float* b1    = (const float*)d_in[6];
    const float* Wpin  = (const float*)d_in[7];
    const float* att   = (const float*)d_in[8];
    const float* Wpost = (const float*)d_in[9];
    const float* bpost = (const float*)d_in[10];
    const float* Wm1   = (const float*)d_in[11];
    const float* bm1   = (const float*)d_in[12];
    const float* Wm2   = (const float*)d_in[13];
    const float* bm2   = (const float*)d_in[14];
    const float* Wm3   = (const float*)d_in[15];
    const float* bm3   = (const float*)d_in[16];
    float* out = (float*)d_out;
    (void)in_sizes; (void)n_in; (void)out_size; (void)ws_size;

    char* ws = (char*)d_ws;
    size_t off = 0;
    auto take = [&](size_t bytes) -> char* {
        char* p = ws + off;
        off = (off + bytes + 255) & ~(size_t)255;
        return p;
    };
    float*     xl    = (float*)take((size_t)NN * CC * 4);
    float*     efeat = (float*)take((size_t)NHE * CC * 4);
    float*     sumb  = (float*)take((size_t)NN * CC * 4);    // -> pnaA halves
    float*     sqb   = (float*)take((size_t)NN * CC * 4);    // -> pnaB halves
    unsigned*  mxe   = (unsigned*)take((size_t)NN * CC * 4); // -> hx overlay
    unsigned*  mne   = (unsigned*)take((size_t)NN * CC * 4);
    float*     araw  = (float*)take((size_t)NNZp * 4);       // -> ex in-place
    float*     ism   = (float*)take((size_t)NN * 4);
    float*     dege  = (float*)take((size_t)NHE * 4);
    float*     natt  = (float*)take((size_t)NN * 4);
    float*     heatt = (float*)take((size_t)NHE * 4);
    unsigned*  menc  = (unsigned*)take((size_t)NN * 4);
    float*     den   = (float*)take((size_t)NN * 4);
    float*     degn  = (float*)take((size_t)NN * 4);
    _Float16*  W1T   = (_Float16*)take((size_t)64 * 32 * 2);   // W1^T  f16
    _Float16*  WT    = (_Float16*)take((size_t)64 * 256 * 2);  // Wpost^T f16 (32KB)
    float*     pooled = (float*)take(128 * 4);
    float*     hx    = (float*)mxe;   // dead after pna_stats; reused for hx

    // ---- init (every call: deterministic, buffers may hold garbage) ----
    fill_f32<<<gridFor((long)NHE * CC, 256), 256, 0, stream>>>(efeat, (long)NHE * CC, 0.0f);
    fill_f32<<<gridFor((long)NN * CC, 256), 256, 0, stream>>>(sumb, (long)NN * CC, 0.0f);
    fill_f32<<<gridFor((long)NN * CC, 256), 256, 0, stream>>>(sqb, (long)NN * CC, 0.0f);
    fill_u32<<<gridFor((long)NN * CC, 256), 256, 0, stream>>>(mxe, (long)NN * CC, 0x007FFFFFu); // enc(-inf)
    fill_u32<<<gridFor((long)NN * CC, 256), 256, 0, stream>>>(mne, (long)NN * CC, 0xFF800000u); // enc(+inf)
    fill_u32<<<gridFor(NN, 256), 256, 0, stream>>>(menc, NN, 0x007FFFFFu);
    fill_f32<<<gridFor(NN, 256), 256, 0, stream>>>(ism, NN, 0.0f);
    fill_f32<<<gridFor(NHE, 256), 256, 0, stream>>>(dege, NHE, 0.0f);
    fill_f32<<<gridFor(NN, 256), 256, 0, stream>>>(den, NN, 0.0f);
    fill_f32<<<gridFor(NN, 256), 256, 0, stream>>>(degn, NN, 0.0f);
    fill_f32<<<1, 128, 0, stream>>>(pooled, 128, 0.0f);

    scatter_ismacro<<<2, 256, 0, stream>>>(midx, ism, 512);
    transpose_w_f16<<<(32 * 64 + 255) / 256, 256, 0, stream>>>(W1, W1T, 32, 64);
    transpose_w_f16<<<(256 * 64 + 255) / 256, 256, 0, stream>>>(Wpost, WT, 256, 64);

    // ---- node embedding GEMM (WMMA) ----
    int ntiles = NN / 16;  // 12500
    xl_gemm<<<(ntiles + 7) / 8, 256, 0, stream>>>(x, fpos, ism, W1T, b1, xl, ntiles);

    // ---- node -> hyperedge aggregation ----
    pin_scatter_he<<<(int)((NNZp * 16 + 255) / 256), 256, 0, stream>>>(ni, ei, pinf, Wpin, xl,
                                                                      efeat, dege);
    he_norm<<<gridFor((long)NHE * CC, 256), 256, 0, stream>>>(efeat, dege);

    // ---- attention logits + per-node softmax ----
    dot_att<<<(NN + 255) / 256, 256, 0, stream>>>(xl, att, 0, natt, NN);
    dot_att<<<(NHE + 255) / 256, 256, 0, stream>>>(efeat, att, 64, heatt, NHE);
    araw_max<<<(int)((NNZp + 255) / 256), 256, 0, stream>>>(ni, ei, natt, heatt, araw, menc);
    decode_m<<<(NN + 255) / 256, 256, 0, stream>>>(menc, NN);
    exp_den<<<(int)((NNZp + 255) / 256), 256, 0, stream>>>(ni, menc, araw, den, degn);

    // ---- hyperedge -> node PNA scatter ----
    pna_scatter<<<(int)((NNZp * 16 + 255) / 256), 256, 0, stream>>>(ni, ei, araw, den, efeat,
                                                                   sumb, sqb, mxe, mne);
    pna_stats<<<(NN + 255) / 256, 256, 0, stream>>>(sumb, sqb, mxe, mne, degn);

    // ---- post GEMM (WMMA, K=256, async-LDS staged weights) + lrelu ----
    hx_gemm<<<(ntiles + 7) / 8, 256, 0, stream>>>(sumb, sqb, WT, bpost, hx, ntiles);

    // ---- pools + MLP head ----
    pool_all<<<512, 256, 0, stream>>>(hx, pooled);
    pool_macro<<<1, 256, 0, stream>>>(hx, midx, pooled);
    mlp_head<<<1, 128, 0, stream>>>(pooled, Wm1, bm1, Wm2, bm2, Wm3, bm3, out);
}
